// StandardCrossAttention_4114578669998
// MI455X (gfx1250) — compile-verified, matrix operand order checked by reading, not run on hardware
//
#include <hip/hip_runtime.h>

// ---------- types ----------
typedef __attribute__((ext_vector_type(16))) __bf16          v16bf;
typedef __attribute__((ext_vector_type(8)))  float           v8f;
typedef __attribute__((ext_vector_type(8)))  unsigned short  us8;
typedef __attribute__((ext_vector_type(4)))  unsigned short  us4;
typedef __attribute__((ext_vector_type(4)))  float           vf4;

union FragB16 { v16bf v; us8 h[2]; };

// ---------- helpers ----------
__device__ __forceinline__ unsigned short f2bf(float f) {
  union { float f; unsigned u; } v; v.f = f;
  unsigned r = v.u + 0x7FFFu + ((v.u >> 16) & 1u);   // round-to-nearest-even
  return (unsigned short)(r >> 16);
}
__device__ __forceinline__ float bf2f(unsigned short h) {
  union { unsigned u; float f; } v; v.u = ((unsigned)h) << 16; return v.f;
}
// A fragment from LDS: two 16B chunks at +0 and +32B (2x ds_load_b128)
__device__ __forceinline__ v16bf frag_from_lds(const unsigned short* p) {
  FragB16 f;
  f.h[0] = *(const us8*)(p);
  f.h[1] = *(const us8*)(p + 16);
  return f.v;
}
// B fragment from pre-packed global: 32 contiguous bytes per lane (2x global_load_b128)
__device__ __forceinline__ v16bf frag_from_packed(const unsigned short* p) {
  FragB16 f;
  f.h[0] = *(const us8*)(p);
  f.h[1] = *(const us8*)(p + 8);
  return f.v;
}
__device__ __forceinline__ v8f wmma_bf16(v16bf a, v16bf b, v8f c) {
  return __builtin_amdgcn_wmma_f32_16x16x32_bf16(false, a, false, b, (short)0, c, false, false);
}
__device__ __forceinline__ float softplusf(float x) {
  if (x > 20.f) return x;
  float e = __expf(x);
  if (x < -20.f) return e;
  return __logf(1.f + e);
}
// stage a 64x256 f32 tile -> bf16 LDS (stride 264 u16), vectorized b128/b64
__device__ __forceinline__ void stage_tile_bf16(const float* __restrict__ src,
                                                unsigned short* dst, int tid) {
  for (int i = tid; i < 4096; i += 256) {
    int r = i >> 6, c4 = (i & 63) << 2;
    vf4 v = *(const vf4*)(src + r * 256 + c4);
    us4 o;
    o[0] = f2bf(v[0]); o[1] = f2bf(v[1]); o[2] = f2bf(v[2]); o[3] = f2bf(v[3]);
    *(us4*)(dst + r * 264 + c4) = o;
  }
}

// ---------- ws layout (bytes) ----------
// [0)        wq_pack  65536 u16
// [131072)   wk_pack  65536 u16
// [262144)   wv_pack  65536 u16
// [393216)   d1_pack  32768 u16
// [458752)   Q        4096*256 f32
// [4653056)  b1       4096*8   f32
#define WS_Q_OFF   458752
#define WS_B1_OFF  4653056

// =====================================================================
// Kernel 1: pack f32 weights into bf16 WMMA-B fragment order.
// packed[((nt*8+ks)*32+lane)*16+e] = bf16(W[ks*32 + perm(e,half)][nt*16+lane%16])
// perm(e,half) = (e<8 ? e : e+8) + 8*half  (ISA 16-bit 32x16 B layout)
// =====================================================================
__global__ void pack_weights(const float* __restrict__ wq,
                             const float* __restrict__ wk,
                             const float* __restrict__ wv,
                             const float* __restrict__ d1,
                             unsigned short* __restrict__ out) {
  int gid = blockIdx.x * 256 + threadIdx.x;
  const float* W; int Ndim; int local; unsigned short* dst;
  if      (gid <  65536) { W = wq; Ndim = 256; local = gid;          dst = out;          }
  else if (gid < 131072) { W = wk; Ndim = 256; local = gid -  65536; dst = out +  65536; }
  else if (gid < 196608) { W = wv; Ndim = 256; local = gid - 131072; dst = out + 131072; }
  else if (gid < 229376) { W = d1; Ndim = 128; local = gid - 196608; dst = out + 196608; }
  else return;
  int e    = local & 15;
  int lane = (local >> 4) & 31;
  int ks   = (local >> 9) & 7;
  int nt   = local >> 12;
  int hlf  = lane >> 4;
  int kk   = ks * 32 + ((e < 8) ? e : e + 8) + hlf * 8;
  int col  = nt * 16 + (lane & 15);
  dst[local] = f2bf(W[kk * Ndim + col]);
}

// =====================================================================
// Kernel 2: Q = h_recv@wq + bq ; b1 = softplus(relu(h_recv@d1+b1)@d2+b2)
// 64 batch rows per block, 8 wave32.
// =====================================================================
__global__ __launch_bounds__(256)
void qmlp_kernel(const float* __restrict__ h_recv,
                 const unsigned short* __restrict__ wq_pack,
                 const float* __restrict__ wq_b,
                 const unsigned short* __restrict__ d1_pack,
                 const float* __restrict__ d1_b,
                 const float* __restrict__ d2_w,
                 const float* __restrict__ d2_b,
                 float* __restrict__ Qout,
                 float* __restrict__ b1out) {
  const int tid = threadIdx.x, wave = tid >> 5, lane = tid & 31;
  const int hlf = lane >> 4, lcol = lane & 15;
  const int row0 = blockIdx.x * 64;

  __shared__ unsigned short hr[64 * 264];   // bf16 h_recv tile
  __shared__ unsigned short Ts[64 * 136];   // bf16 relu output

  stage_tile_bf16(h_recv + (long)row0 * 256, hr, tid);
  __syncthreads();

  // Q: two 16-col tiles per wave (16 tiles over 256 cols)
  #pragma unroll 1
  for (int j = 0; j < 2; ++j) {
    int nt = wave * 2 + j;
    v8f acc[4] = {};
    #pragma unroll 1
    for (int ks = 0; ks < 8; ++ks) {
      v16bf bq = frag_from_packed(wq_pack + (((nt * 8 + ks) * 32 + lane) << 4));
      #pragma unroll
      for (int m = 0; m < 4; ++m) {
        v16bf a = frag_from_lds(hr + (m * 16 + lcol) * 264 + ks * 32 + hlf * 8);
        acc[m] = wmma_bf16(a, bq, acc[m]);
      }
    }
    int col = nt * 16 + lcol;
    float bias = wq_b[col];
    for (int m = 0; m < 4; ++m)
      for (int r = 0; r < 8; ++r)
        Qout[(row0 + m * 16 + r + hlf * 8) * 256 + col] = acc[m][r] + bias;
  }

  // T = relu(h_recv@d1 + b): one 16-col tile per wave (8 tiles over 128 cols)
  {
    int nt = wave;
    v8f acc[4] = {};
    #pragma unroll 1
    for (int ks = 0; ks < 8; ++ks) {
      v16bf bd = frag_from_packed(d1_pack + (((nt * 8 + ks) * 32 + lane) << 4));
      #pragma unroll
      for (int m = 0; m < 4; ++m) {
        v16bf a = frag_from_lds(hr + (m * 16 + lcol) * 264 + ks * 32 + hlf * 8);
        acc[m] = wmma_bf16(a, bd, acc[m]);
      }
    }
    int col = nt * 16 + lcol;
    float bias = d1_b[col];
    for (int m = 0; m < 4; ++m)
      for (int r = 0; r < 8; ++r) {
        float t = acc[m][r] + bias;
        Ts[(m * 16 + r + hlf * 8) * 136 + col] = f2bf(t > 0.f ? t : 0.f);
      }
  }
  __syncthreads();

  // b1 = softplus(T @ d2 + b2)   (512 dots of length 128)
  for (int p = tid; p < 512; p += 256) {
    int r = p >> 3, h = p & 7;
    float s = d2_b[h];
    for (int k = 0; k < 128; ++k)
      s += bf2f(Ts[r * 136 + k]) * d2_w[k * 8 + h];
    b1out[(row0 + r) * 8 + h] = softplusf(s);
  }
}

// =====================================================================
// Kernel 3: per-batch fused K/V projection + scores + softmax + z.
// One block per batch; 8 wave32; wave w owns head w's 32 output cols.
// ks-outer loop shares each A fragment across both column tiles and
// both K/V accumulators (4 WMMAs per A-fragment load).
// =====================================================================
__global__ __launch_bounds__(256)
void attn_kernel(const float* __restrict__ h_send,
                 const float* __restrict__ dist,
                 const int*   __restrict__ mask,
                 const unsigned short* __restrict__ wk_pack,
                 const unsigned short* __restrict__ wv_pack,
                 const float* __restrict__ wk_b,
                 const float* __restrict__ wv_b,
                 const float* __restrict__ Qin,
                 const float* __restrict__ b1in,
                 float* __restrict__ z_out,
                 float* __restrict__ attn_out) {
  const int b = blockIdx.x;
  const int tid = threadIdx.x, wave = tid >> 5, lane = tid & 31;
  const int hlf = lane >> 4, lcol = lane & 15;

  __shared__ unsigned short hs[64 * 264];   // bf16 h_send[b]
  __shared__ unsigned short Ks[64 * 264];   // bf16 K, row-major [n][c]
  __shared__ unsigned short Vt[256 * 72];   // bf16 V, transposed [c][n]
  __shared__ float sc[64 * 8];              // scores [n][h]
  __shared__ float atT[8 * 64];             // attn   [h][n]
  __shared__ float qsh[256];
  __shared__ float b1sh[8];

  // ---- stage h_send[b] (HBM stream, read exactly once) + Q/b1 ----
  stage_tile_bf16(h_send + (long)b * 64 * 256, hs, tid);
  qsh[tid] = Qin[b * 256 + tid];
  if (tid < 8) b1sh[tid] = b1in[b * 8 + tid];
  __syncthreads();

  // ---- K and V projections ----
  {
    v8f aK[2][4] = {};
    v8f aV[2][4] = {};
    #pragma unroll 1
    for (int ks = 0; ks < 8; ++ks) {
      v16bf a[4];
      #pragma unroll
      for (int m = 0; m < 4; ++m)
        a[m] = frag_from_lds(hs + (m * 16 + lcol) * 264 + ks * 32 + hlf * 8);
      #pragma unroll
      for (int j = 0; j < 2; ++j) {
        int nt = wave * 2 + j;
        int po = ((nt * 8 + ks) * 32 + lane) << 4;
        v16bf bK = frag_from_packed(wk_pack + po);
        v16bf bV = frag_from_packed(wv_pack + po);
        #pragma unroll
        for (int m = 0; m < 4; ++m) {
          aK[j][m] = wmma_bf16(a[m], bK, aK[j][m]);
          aV[j][m] = wmma_bf16(a[m], bV, aV[j][m]);
        }
      }
    }
    // epilogue: bias add; K row-major (scalar b16), V transposed (packed b128)
    #pragma unroll
    for (int j = 0; j < 2; ++j) {
      int col = (wave * 2 + j) * 16 + lcol;
      float bk = wk_b[col], bv = wv_b[col];
      #pragma unroll
      for (int m = 0; m < 4; ++m) {
        us8 vv;
        #pragma unroll
        for (int r = 0; r < 8; ++r) {
          int row = m * 16 + r + hlf * 8;           // C layout: C[r+8*half][lane%16]
          Ks[row * 264 + col] = f2bf(aK[j][m][r] + bk);
          vv[r] = f2bf(aV[j][m][r] + bv);
        }
        *(us8*)(Vt + col * 72 + m * 16 + hlf * 8) = vv;  // rows contiguous
      }
    }
  }
  __syncthreads();

  // ---- scores[n,h] = Q[h]·K[n,h]/sqrt(32) - b1[h]*dist[n] ; mask ----
  for (int p = tid; p < 512; p += 256) {
    int n = p >> 3, h = p & 7;
    const us8* kr = (const us8*)(Ks + n * 264 + h * 32);
    const vf4* qr = (const vf4*)(qsh + h * 32);
    float s = 0.f;
    #pragma unroll
    for (int ch = 0; ch < 4; ++ch) {
      us8 k8 = kr[ch];
      vf4 q0 = qr[2 * ch], q1 = qr[2 * ch + 1];
      s += q0[0] * bf2f(k8[0]) + q0[1] * bf2f(k8[1]) +
           q0[2] * bf2f(k8[2]) + q0[3] * bf2f(k8[3]) +
           q1[0] * bf2f(k8[4]) + q1[1] * bf2f(k8[5]) +
           q1[2] * bf2f(k8[6]) + q1[3] * bf2f(k8[7]);
    }
    s *= 0.17677669529663687f;                    // 1/sqrt(32)
    s -= b1sh[h] * dist[b * 64 + n];
    if (mask[b * 64 + n] == 0) s = -__builtin_inff();
    sc[n * 8 + h] = s;
  }
  __syncthreads();

  // ---- softmax over n (64) per head: wave w == head w, wave32 shuffles ----
  {
    int h = wave;
    float v0 = sc[lane * 8 + h];
    float v1 = sc[(lane + 32) * 8 + h];
    float mx = fmaxf(v0, v1);
    for (int off = 16; off > 0; off >>= 1) mx = fmaxf(mx, __shfl_xor(mx, off, 32));
    float e0 = __expf(v0 - mx), e1 = __expf(v1 - mx);
    float s = e0 + e1;
    for (int off = 16; off > 0; off >>= 1) s += __shfl_xor(s, off, 32);
    float inv = 1.f / s;
    float a0 = e0 * inv, a1 = e1 * inv;
    atT[h * 64 + lane] = a0;
    atT[h * 64 + lane + 32] = a1;
    attn_out[(long)b * 512 + lane * 8 + h] = a0;            // [B,N,H]
    attn_out[(long)b * 512 + (lane + 32) * 8 + h] = a1;
  }
  __syncthreads();

  // ---- z[c] = sum_n attn[n, c/32] * V[n, c]  (vectorized over n) ----
  {
    int c = tid, h = c >> 5;
    const us8* vr = (const us8*)(Vt + c * 72);
    const vf4* ar = (const vf4*)(atT + h * 64);
    float z = 0.f;
    #pragma unroll
    for (int ch = 0; ch < 8; ++ch) {
      us8 v8 = vr[ch];
      vf4 a0 = ar[2 * ch], a1 = ar[2 * ch + 1];
      z += a0[0] * bf2f(v8[0]) + a0[1] * bf2f(v8[1]) +
           a0[2] * bf2f(v8[2]) + a0[3] * bf2f(v8[3]) +
           a1[0] * bf2f(v8[4]) + a1[1] * bf2f(v8[5]) +
           a1[2] * bf2f(v8[6]) + a1[3] * bf2f(v8[7]);
    }
    z_out[(long)b * 256 + c] = z;
  }
}

// =====================================================================
extern "C" void kernel_launch(void* const* d_in, const int* in_sizes, int n_in,
                              void* d_out, int out_size, void* d_ws, size_t ws_size,
                              hipStream_t stream) {
  (void)in_sizes; (void)n_in; (void)out_size; (void)ws_size;
  const float* h_recv = (const float*)d_in[0];
  const float* h_send = (const float*)d_in[1];
  const float* dist   = (const float*)d_in[2];
  const int*   mask   = (const int*)  d_in[3];
  const float* wq_w = (const float*)d_in[4];
  const float* wq_b = (const float*)d_in[5];
  const float* wk_w = (const float*)d_in[6];
  const float* wk_b = (const float*)d_in[7];
  const float* wv_w = (const float*)d_in[8];
  const float* wv_b = (const float*)d_in[9];
  const float* d1_w = (const float*)d_in[10];
  const float* d1_b = (const float*)d_in[11];
  const float* d2_w = (const float*)d_in[12];
  const float* d2_b = (const float*)d_in[13];

  unsigned short* packs = (unsigned short*)d_ws;
  float* Qws   = (float*)((char*)d_ws + WS_Q_OFF);
  float* b1ws  = (float*)((char*)d_ws + WS_B1_OFF);
  float* z_out = (float*)d_out;
  float* attn_out = z_out + 4096 * 256;

  pack_weights<<<896, 256, 0, stream>>>(wq_w, wk_w, wv_w, d1_w, packs);
  qmlp_kernel<<<64, 256, 0, stream>>>(h_recv, packs, wq_b, packs + 196608, d1_b,
                                      d2_w, d2_b, Qws, b1ws);
  attn_kernel<<<4096, 256, 0, stream>>>(h_send, dist, mask,
                                        packs + 65536, packs + 131072,
                                        wk_b, wv_b, Qws, b1ws, z_out, attn_out);
}
